// GAT2_72181220377182
// MI455X (gfx1250) — matrix operand order repaired and checked by reading, not compiled, not run on hardware
//
#include <hip/hip_runtime.h>
#include <float.h>

#define TPB 256

typedef __attribute__((ext_vector_type(16))) _Float16 v16h;
typedef __attribute__((ext_vector_type(8)))  float    v8f;

// ---------- helpers ----------

// Float atomic max via signed/unsigned integer atomics (safe with -FLT_MAX init)
__device__ __forceinline__ void atomicMaxF(float* addr, float val) {
  if (val >= 0.0f) atomicMax((int*)addr, __float_as_int(val));
  else             atomicMin((unsigned int*)addr, __float_as_uint(val));
}

__global__ void k_fill_f32(float* __restrict__ p, long long n, float v) {
  long long t = (long long)blockIdx.x * TPB + threadIdx.x;
  if (t < n) p[t] = v;
}

__global__ void k_cvt_f16(const float* __restrict__ x, _Float16* __restrict__ y, long long n) {
  long long t = (long long)blockIdx.x * TPB + threadIdx.x;
  if (t < n) y[t] = (_Float16)x[t];
}

// Pack row-major f32 W[K][Ncols] into WMMA B-operand layout:
// flat = ((nt*Ktiles + kt)*32 + lane)*16 + j ;  per ISA: lane holds column n=lane&15,
// K = kt*32 + (j>>3)*16 + (lane>>4)*8 + (j&7)  (dense 16-bit B 32x16 layout)
__global__ void k_pack_w(const float* __restrict__ W, _Float16* __restrict__ out, int Ncols, int K) {
  int t = blockIdx.x * TPB + threadIdx.x;
  int total = K * Ncols;
  if (t >= total) return;
  int Ktiles = K >> 5;
  int j    = t & 15;
  int lane = (t >> 4) & 31;
  int kt   = (t >> 9) % Ktiles;
  int nt   = t / (512 * Ktiles);
  int k = kt * 32 + (j >> 3) * 16 + (lane >> 4) * 8 + (j & 7);
  int n = nt * 16 + (lane & 15);
  out[t] = (_Float16)W[k * Ncols + n];
}

// C[M][Ncols] f32 = A[M][K] f16 (row-major) * Bp (packed).  One wave per 16x16 tile.
// A 16x32 f16 layout: lanes0-15 M=0..15 hold K {0..7,16..23}; lanes16-31 hold K {8..15,24..31}.
// C 16x16 f32 layout: lane -> N = lane&15, element r -> M = 8*(lane>>4) + r.
__global__ void k_gemm_wmma(const _Float16* __restrict__ A, const _Float16* __restrict__ Bp,
                            float* __restrict__ C, int M, int K, int Ncols) {
  int wave = (int)((blockIdx.x * blockDim.x + threadIdx.x) >> 5);
  int lane = threadIdx.x & 31;
  int Ntiles = Ncols >> 4;
  int Mtiles = M >> 4;
  int Ktiles = K >> 5;
  if (wave >= Mtiles * Ntiles) return;        // wave-uniform guard: EXEC stays all-1s
  int mt = wave / Ntiles, nt = wave - mt * Ntiles;
  int l15 = lane & 15, khalf = lane >> 4;
  union { v8f v; float e[8]; } c;
#pragma unroll
  for (int i = 0; i < 8; ++i) c.e[i] = 0.0f;
  for (int kt = 0; kt < Ktiles; ++kt) {
    const _Float16* arow = A + (size_t)(mt * 16 + l15) * K + kt * 32;
    union { v16h v; _Float16 e[16]; } a;
#pragma unroll
    for (int i = 0; i < 8; ++i) {
      a.e[i]     = arow[khalf * 8 + i];        // K = khalf*8 + i
      a.e[8 + i] = arow[16 + khalf * 8 + i];   // K = 16 + khalf*8 + i
    }
    v16h b = *(const v16h*)(Bp + (((size_t)(nt * Ktiles + kt)) << 9) + ((size_t)lane << 4));
    c.v = __builtin_amdgcn_wmma_f32_16x16x32_f16(false, a.v, false, b, (short)0, c.v,
                                                 false, false);
  }
#pragma unroll
  for (int r = 0; r < 8; ++r)
    C[(size_t)(mt * 16 + khalf * 8 + r) * Ncols + nt * 16 + l15] = c.e[r];
}

// el[n,h] = <h1[n,h,:], al[h,:]> ; er likewise
__global__ void k_node_logits(const float* __restrict__ h, const float* __restrict__ al,
                              const float* __restrict__ ar, float* __restrict__ el,
                              float* __restrict__ er, int N, int H, int D) {
  int t = blockIdx.x * TPB + threadIdx.x;
  if (t >= N * H) return;
  int n = t / H, hh = t - n * H;
  const float* row = h + (size_t)n * H * D + (size_t)hh * D;
  const float* alr = al + hh * D;
  const float* arr = ar + hh * D;
  float sl = 0.f, sr = 0.f;
  for (int d = 0; d < D; ++d) { float v = row[d]; sl += v * alr[d]; sr += v * arr[d]; }
  el[t] = sl; er[t] = sr;
}

// segment_max of leaky-relu logits over dst
__global__ void k_edge_max(const int* __restrict__ src, const int* __restrict__ dst,
                           const float* __restrict__ el, const float* __restrict__ er,
                           float* __restrict__ emax, int E, int H) {
  int t = blockIdx.x * TPB + threadIdx.x;
  if (t >= E * H) return;
  int e = t / H, hh = t - e * H;
  int dn = dst[e] * H + hh;
  float v = el[src[e] * H + hh] + er[dn];
  v = v > 0.f ? v : 0.2f * v;
  atomicMaxF(&emax[dn], v);
}

// ex = exp(logit - max); denom[dst] += ex; stash ex per (edge, head)
__global__ void k_edge_exp(const int* __restrict__ src, const int* __restrict__ dst,
                           const float* __restrict__ el, const float* __restrict__ er,
                           const float* __restrict__ emax, float* __restrict__ exbuf,
                           float* __restrict__ denom, int E, int H) {
  int t = blockIdx.x * TPB + threadIdx.x;
  if (t >= E * H) return;
  int e = t / H, hh = t - e * H;
  int dn = dst[e] * H + hh;
  float v = el[src[e] * H + hh] + er[dn];
  v = v > 0.f ? v : 0.2f * v;
  float ex = __expf(v - emax[dn]);
  exbuf[t] = ex;
  atomicAdd(&denom[dn], ex);
}

// msg[dst, h, d] += ex[e,h] * h1[src, h, d]   (d fastest -> coalesced)
__global__ void k_edge_aggr(const int* __restrict__ src, const int* __restrict__ dst,
                            const float* __restrict__ exbuf, const float* __restrict__ h,
                            float* __restrict__ msg, long long total, int H, int D) {
  long long t = (long long)blockIdx.x * TPB + threadIdx.x;
  if (t >= total) return;
  int HD = H * D;
  int e = (int)(t / HD);
  int r = (int)(t - (long long)e * HD);
  float w = exbuf[e * H + r / D];
  atomicAdd(&msg[(size_t)dst[e] * HD + r], w * h[(size_t)src[e] * HD + r]);
}

// out[n,d] = sum_h relu(msg[n,h,d]/denom[n,h] + bias[h,d]) ; write f32 and f16 copies
__global__ void k_finalize(const float* __restrict__ msg, const float* __restrict__ denom,
                           const float* __restrict__ bias, float* __restrict__ outf,
                           _Float16* __restrict__ outh, int N, int H, int D) {
  int t = blockIdx.x * TPB + threadIdx.x;
  if (t >= N * D) return;
  int n = t / D, d = t - n * D;
  float s = 0.f;
  for (int hh = 0; hh < H; ++hh) {
    float den = denom[n * H + hh];
    float v = (den > 0.f) ? msg[((size_t)n * H + hh) * D + d] / den : 0.f;
    v += bias[hh * D + d];
    s += v > 0.f ? v : 0.f;
  }
  outf[t] = s;
  outh[t] = (_Float16)s;
}

__global__ void k_readout_max(const float* __restrict__ hf, const int* __restrict__ gid,
                              float* __restrict__ ro, int N, int D) {
  int t = blockIdx.x * TPB + threadIdx.x;
  if (t >= N * D) return;
  int n = t / D, d = t - n * D;
  atomicMaxF(&ro[gid[n] * D + d], hf[t]);
}

// tiny head MLP: [32,256] @ W1[256,128] + b1 -> relu -> @ W2[128,1] + b2 -> relu
__global__ void k_mlp(const float* __restrict__ rlig, const float* __restrict__ rrec,
                      const float* __restrict__ W1, const float* __restrict__ b1,
                      const float* __restrict__ W2, const float* __restrict__ b2,
                      float* __restrict__ out) {
  __shared__ float t1[32][128];
  int d = threadIdx.x;  // 0..127
  for (int b = 0; b < 32; ++b) {
    float acc = b1[d];
    for (int k = 0; k < 128; ++k) acc += rlig[b * 128 + k] * W1[k * 128 + d];
    for (int k = 0; k < 128; ++k) acc += rrec[b * 128 + k] * W1[(128 + k) * 128 + d];
    t1[b][d] = acc > 0.f ? acc : 0.f;
  }
  __syncthreads();
  if (d < 32) {
    float acc = b2[0];
    for (int k = 0; k < 128; ++k) acc += t1[d][k] * W2[k];
    out[d] = acc > 0.f ? acc : 0.f;
  }
}

// ---------- host side ----------

static inline int nblk(long long n) { return (int)((n + TPB - 1) / TPB); }

static void run_branch(const float* x, const int* src, const int* dst, const int* gid,
                       const float* W1, const float* al1, const float* ar1, const float* b1,
                       const float* W2, const float* al2, const float* ar2, const float* b2,
                       int N, int E,
                       _Float16* xh, _Float16* w1p, _Float16* w2p,
                       float* h1, float* msg, float* el, float* er,
                       float* emax, float* denom, float* exbuf,
                       _Float16* hsum16, float* hout, float* ro, hipStream_t stream) {
  const int H1 = 10, D1 = 64, HD1 = 640, D2 = 128, K = 64;

  k_pack_w<<<nblk((long long)K * HD1), TPB, 0, stream>>>(W1, w1p, HD1, K);
  k_pack_w<<<nblk((long long)K * D2),  TPB, 0, stream>>>(W2, w2p, D2, K);
  k_cvt_f16<<<nblk((long long)N * K), TPB, 0, stream>>>(x, xh, (long long)N * K);

  // ---- layer 1 ----
  long long tiles1 = (long long)(N / 16) * (HD1 / 16);
  k_gemm_wmma<<<nblk(tiles1 * 32), TPB, 0, stream>>>(xh, w1p, h1, N, K, HD1);
  k_node_logits<<<nblk((long long)N * H1), TPB, 0, stream>>>(h1, al1, ar1, el, er, N, H1, D1);
  k_fill_f32<<<nblk((long long)N * H1), TPB, 0, stream>>>(emax, (long long)N * H1, -FLT_MAX);
  k_fill_f32<<<nblk((long long)N * H1), TPB, 0, stream>>>(denom, (long long)N * H1, 0.f);
  k_edge_max<<<nblk((long long)E * H1), TPB, 0, stream>>>(src, dst, el, er, emax, E, H1);
  k_edge_exp<<<nblk((long long)E * H1), TPB, 0, stream>>>(src, dst, el, er, emax, exbuf, denom, E, H1);
  k_fill_f32<<<nblk((long long)N * HD1), TPB, 0, stream>>>(msg, (long long)N * HD1, 0.f);
  k_edge_aggr<<<nblk((long long)E * HD1), TPB, 0, stream>>>(src, dst, exbuf, h1, msg,
                                                            (long long)E * HD1, H1, D1);
  k_finalize<<<nblk((long long)N * D1), TPB, 0, stream>>>(msg, denom, b1, hout, hsum16, N, H1, D1);

  // ---- layer 2 (reuse h1 as h2, same edge scratch) ----
  long long tiles2 = (long long)(N / 16) * (D2 / 16);
  k_gemm_wmma<<<nblk(tiles2 * 32), TPB, 0, stream>>>(hsum16, w2p, h1, N, K, D2);
  k_node_logits<<<nblk((long long)N), TPB, 0, stream>>>(h1, al2, ar2, el, er, N, 1, D2);
  k_fill_f32<<<nblk((long long)N), TPB, 0, stream>>>(emax, (long long)N, -FLT_MAX);
  k_fill_f32<<<nblk((long long)N), TPB, 0, stream>>>(denom, (long long)N, 0.f);
  k_edge_max<<<nblk((long long)E), TPB, 0, stream>>>(src, dst, el, er, emax, E, 1);
  k_edge_exp<<<nblk((long long)E), TPB, 0, stream>>>(src, dst, el, er, emax, exbuf, denom, E, 1);
  k_fill_f32<<<nblk((long long)N * D2), TPB, 0, stream>>>(msg, (long long)N * D2, 0.f);
  k_edge_aggr<<<nblk((long long)E * D2), TPB, 0, stream>>>(src, dst, exbuf, h1, msg,
                                                           (long long)E * D2, 1, D2);
  k_finalize<<<nblk((long long)N * D2), TPB, 0, stream>>>(msg, denom, b2, hout, hsum16, N, 1, D2);
  k_readout_max<<<nblk((long long)N * D2), TPB, 0, stream>>>(hout, gid, ro, N, D2);
}

extern "C" void kernel_launch(void* const* d_in, const int* in_sizes, int n_in,
                              void* d_out, int out_size, void* d_ws, size_t ws_size,
                              hipStream_t stream) {
  (void)in_sizes; (void)n_in; (void)out_size; (void)ws_size;
  const int N_LIG = 10000, E_LIG = 80000, N_REC = 30000, E_REC = 480000;

  const float* x_lig  = (const float*)d_in[0];
  const int*   s_lig  = (const int*)d_in[1];
  const int*   t_lig  = (const int*)d_in[2];
  const int*   g_lig  = (const int*)d_in[3];
  const float* x_rec  = (const float*)d_in[4];
  const int*   s_rec  = (const int*)d_in[5];
  const int*   t_rec  = (const int*)d_in[6];
  const int*   g_rec  = (const int*)d_in[7];
  const float* W1l = (const float*)d_in[8],  *al1l = (const float*)d_in[9];
  const float* ar1l = (const float*)d_in[10], *b1l = (const float*)d_in[11];
  const float* W2l = (const float*)d_in[12], *al2l = (const float*)d_in[13];
  const float* ar2l = (const float*)d_in[14], *b2l = (const float*)d_in[15];
  const float* W1r = (const float*)d_in[16], *al1r = (const float*)d_in[17];
  const float* ar1r = (const float*)d_in[18], *b1r = (const float*)d_in[19];
  const float* W2r = (const float*)d_in[20], *al2r = (const float*)d_in[21];
  const float* ar2r = (const float*)d_in[22], *b2r = (const float*)d_in[23];
  const float* Wlin1 = (const float*)d_in[24], *blin1 = (const float*)d_in[25];
  const float* Wlin2 = (const float*)d_in[26], *blin2 = (const float*)d_in[27];

  // carve scratch (sized for the rec branch; branches run sequentially and reuse it)
  char* p = (char*)d_ws;
  auto carve = [&p](size_t bytes) -> char* {
    char* r = p;
    p += (bytes + 255) & ~(size_t)255;
    return r;
  };
  const int NM = N_REC, EM = E_REC;
  _Float16* xh    = (_Float16*)carve((size_t)NM * 64 * 2);
  _Float16* w1p   = (_Float16*)carve((size_t)64 * 640 * 2);
  _Float16* w2p   = (_Float16*)carve((size_t)64 * 128 * 2);
  float* h1       = (float*)carve((size_t)NM * 640 * 4);
  float* msg      = (float*)carve((size_t)NM * 640 * 4);
  float* el       = (float*)carve((size_t)NM * 10 * 4);
  float* er       = (float*)carve((size_t)NM * 10 * 4);
  float* emax     = (float*)carve((size_t)NM * 10 * 4);
  float* denom    = (float*)carve((size_t)NM * 10 * 4);
  float* exbuf    = (float*)carve((size_t)EM * 10 * 4);
  _Float16* hs16  = (_Float16*)carve((size_t)NM * 128 * 2);
  float* hout     = (float*)carve((size_t)NM * 128 * 4);
  float* rlig     = (float*)carve((size_t)32 * 128 * 4);
  float* rrec     = (float*)carve((size_t)32 * 128 * 4);

  k_fill_f32<<<nblk(32 * 128), TPB, 0, stream>>>(rlig, 32 * 128, -FLT_MAX);
  k_fill_f32<<<nblk(32 * 128), TPB, 0, stream>>>(rrec, 32 * 128, -FLT_MAX);

  run_branch(x_lig, s_lig, t_lig, g_lig, W1l, al1l, ar1l, b1l, W2l, al2l, ar2l, b2l,
             N_LIG, E_LIG, xh, w1p, w2p, h1, msg, el, er, emax, denom, exbuf,
             hs16, hout, rlig, stream);
  run_branch(x_rec, s_rec, t_rec, g_rec, W1r, al1r, ar1r, b1r, W2r, al2r, ar2r, b2r,
             N_REC, E_REC, xh, w1p, w2p, h1, msg, el, er, emax, denom, exbuf,
             hs16, hout, rrec, stream);

  k_mlp<<<1, 128, 0, stream>>>(rlig, rrec, Wlin1, blin1, Wlin2, blin2, (float*)d_out);
}
